// PointMultiGraspNet_V3_90658169683991
// MI455X (gfx1250) — matrix operand order repaired and checked by reading, not compiled, not run on hardware
//
#include <hip/hip_runtime.h>

typedef __attribute__((ext_vector_type(16))) _Float16 v16h;
typedef __attribute__((ext_vector_type(8)))  float    v8f;

#define BATCH 1024
#define SEGK  512
#define CH    512
#define DIN   544     // CH + 32
#define HID   256
#define KCLS  48
#define LN_EPS 1e-5f

// ---------------------------------------------------------------------------
// Kernel 1: ragged segment max (memory-bound, ~1 GiB streamed @ 23.3 TB/s).
// One block per segment; 256 threads split into two 128-lane halves that
// process interleaved rows (even/odd). Each lane reads float4 (b128), so one
// half covers a full 2KB row per clause; x4 row unroll => 4 b128 loads in
// flight per thread. Halves combined through LDS at the end.
// ---------------------------------------------------------------------------
__global__ __launch_bounds__(256)
void segmax_kernel(const float* __restrict__ feat,
                   const long long* __restrict__ offsets,
                   float* __restrict__ features_out,   // [BATCH][CH] (d_out)
                   float* __restrict__ x)              // [BATCH][DIN] (ws)
{
    __shared__ float4 red[256];

    const int b    = blockIdx.x;
    const int tid  = threadIdx.x;
    const int half = tid >> 7;          // 0 or 1: even/odd rows
    const int l128 = tid & 127;         // lane within half: 4 channels each
    const long long start = (b == 0) ? 0LL : offsets[b - 1];
    const long long end   = offsets[b];

    const float NEG = -__builtin_inff();
    float4 acc = make_float4(NEG, NEG, NEG, NEG);

    const float4* rowp = (const float4*)(feat + (start + half) * CH) + l128;
    const int strideV = 2 * (CH / 4);   // float4 units between my rows

    long long r = start + half;
    for (; r + 8 <= end; r += 8) {      // 4 of my rows per iter
        float4 v0 = rowp[0];
        float4 v1 = rowp[strideV];
        float4 v2 = rowp[2 * strideV];
        float4 v3 = rowp[3 * strideV];
        __builtin_prefetch(rowp + 8 * strideV, 0, 0);
        acc.x = fmaxf(acc.x, fmaxf(fmaxf(v0.x, v1.x), fmaxf(v2.x, v3.x)));
        acc.y = fmaxf(acc.y, fmaxf(fmaxf(v0.y, v1.y), fmaxf(v2.y, v3.y)));
        acc.z = fmaxf(acc.z, fmaxf(fmaxf(v0.z, v1.z), fmaxf(v2.z, v3.z)));
        acc.w = fmaxf(acc.w, fmaxf(fmaxf(v0.w, v1.w), fmaxf(v2.w, v3.w)));
        rowp += 4 * strideV;
    }
    for (; r < end; r += 2) {
        float4 v = rowp[0];
        acc.x = fmaxf(acc.x, v.x);
        acc.y = fmaxf(acc.y, v.y);
        acc.z = fmaxf(acc.z, v.z);
        acc.w = fmaxf(acc.w, v.w);
        rowp += strideV;
    }

    red[tid] = acc;
    __syncthreads();

    if (tid < 128) {
        float4 a0 = red[tid];
        float4 a1 = red[tid + 128];
        float4 m = make_float4(fmaxf(a0.x, a1.x), fmaxf(a0.y, a1.y),
                               fmaxf(a0.z, a1.z), fmaxf(a0.w, a1.w));
        // torch pads with zeros before max when segment shorter than SEGK
        // (empty segment -> -inf -> clamped to 0, matching the reference).
        if (end - start < (long long)SEGK) {
            m.x = fmaxf(m.x, 0.0f); m.y = fmaxf(m.y, 0.0f);
            m.z = fmaxf(m.z, 0.0f); m.w = fmaxf(m.w, 0.0f);
        }
        *(float4*)(features_out + b * CH + 4 * tid) = m;
        *(float4*)(x + b * DIN + 4 * tid) = m;   // DIN multiple of 4: aligned
    }
}

// ---------------------------------------------------------------------------
// Kernel 2: info features -> x[:, 512:544].  K=3 GEMM, trivial.
// ---------------------------------------------------------------------------
__global__ __launch_bounds__(256)
void info_kernel(const float* __restrict__ info,
                 const float* __restrict__ w_info,
                 const float* __restrict__ b_info,
                 float* __restrict__ x)
{
    int tid = blockIdx.x * blockDim.x + threadIdx.x;
    if (tid >= BATCH * 32) return;
    int b = tid >> 5, j = tid & 31;
    float s = b_info[j];
    #pragma unroll
    for (int k = 0; k < 3; ++k)
        s += info[b * 3 + k] * w_info[k * 32 + j];
    x[b * DIN + CH + j] = s;
}

// ---------------------------------------------------------------------------
// Kernel 3: one-time weight f32->f16 conversion + swizzle into the exact
// per-lane WMMA B-matrix layout, so the head kernel loads one contiguous
// v16h (32 B) per lane per WMMA step.
// Layout: out[((kt*NT + ct)*32 + lane)*16 + e] = w[K][col],
//   K = kt*32 + (e&7) + 16*(e>>3) + 8*(lane>>4),  col = ct*16 + (lane&15).
// ---------------------------------------------------------------------------
__global__ __launch_bounds__(256)
void swizzle_b_kernel(const float* __restrict__ w, _Float16* __restrict__ out,
                      int Ktot, int N)
{
    int idx = blockIdx.x * blockDim.x + threadIdx.x;
    if (idx >= Ktot * N) return;
    int e    = idx & 15;
    int lane = (idx >> 4) & 31;
    int rest = idx >> 9;
    int NT   = N >> 4;
    int ct   = rest % NT;
    int kt   = rest / NT;
    int K    = kt * 32 + (e & 7) + ((e >> 3) << 4) + ((lane >> 4) << 3);
    int col  = (ct << 4) + (lane & 15);
    out[idx] = (_Float16)w[K * N + col];
}

// ---------------------------------------------------------------------------
// Kernel 4: fused MLP head. One block handles 16 rows:
//   GEMM1 (x[16,544] @ w1[544,256]) via v_wmma_f32_16x16x32_f16 (f32 accum)
//   -> LayerNorm(256) -> ReLU -> GEMM2 (@ w2[256,n2]) via WMMA -> +bias.
// A operands pre-swizzled in LDS (2x ds_load_b128/lane/step); B operands
// pre-swizzled in global f16 (2x global_load_b128/lane/step, L2-resident).
// 256 threads = 8 waves; one wave per 16x16 output tile; WMMA loops
// wave-uniform so EXEC stays all-ones.
// ---------------------------------------------------------------------------
#define KT1 (DIN / 32)    // 17
#define KT2 (HID / 32)    // 8

__global__ __launch_bounds__(256)
void head_kernel(const float* __restrict__ x,      // [BATCH][DIN]
                 const _Float16* __restrict__ w1s, // swizzled [KT1*16*32*16]
                 const float* __restrict__ b1,     // [HID]
                 const float* __restrict__ g,      // [HID]
                 const float* __restrict__ be,     // [HID]
                 const _Float16* __restrict__ w2s, // swizzled [KT2*(n2/16)*32*16]
                 const float* __restrict__ b2,     // [n2]
                 float* __restrict__ out,          // [BATCH][n2]
                 int n2)
{
    __shared__ _Float16 aS[KT1 * 32 * 16];   // 17408 B : x tile, A-layout
    __shared__ float    hf[16 * HID];        // 16384 B : GEMM1 result (f32)
    __shared__ _Float16 h16s[KT2 * 32 * 16]; //  8192 B : post-LN act, A-layout

    const int tid  = threadIdx.x;
    const int row0 = blockIdx.x * 16;

    // Stage x tile into LDS as f16, pre-swizzled to WMMA A layout:
    // aS[(kk*32 + lane)*16 + e] = x[row0 + (lane&15)][Ka],
    //   Ka = kk*32 + (e&7) + (e>>3)*16 + 8*(lane>>4).
    for (int i = tid; i < KT1 * 32 * 16; i += 256) {
        int e  = i & 15;
        int ls = (i >> 4) & 31;
        int kk = i >> 9;
        int m  = ls & 15, kg = ls >> 4;
        int K  = kk * 32 + (e & 7) + ((e >> 3) << 4) + (kg << 3);
        aS[i] = (_Float16)x[(row0 + m) * DIN + K];
    }
    __syncthreads();

    const int wave = tid >> 5;       // wave-uniform
    const int lane = tid & 31;
    const int mn   = lane & 15;      // D-matrix N index
    const int kg   = lane >> 4;      // lane group (M split in D layout)

    const v16h* bW1 = (const v16h*)w1s;

    // ---- GEMM1: h = x @ w1 + b1 ; 16 column tiles, 2 per wave --------------
    for (int tcol = wave; tcol < HID / 16; tcol += 8) {
        v8f acc = {};
        #pragma unroll
        for (int kk = 0; kk < KT1; ++kk) {
            v16h a  = *(const v16h*)&aS[(kk * 32 + lane) * 16];
            v16h bm = bW1[(kk * (HID / 16) + tcol) * 32 + lane];
            acc = __builtin_amdgcn_wmma_f32_16x16x32_f16(
                false, a, false, bm, (short)0, acc, false, false);
        }
        const int col0 = tcol * 16;
        #pragma unroll
        for (int r = 0; r < 8; ++r) {               // D: M = r + 8*kg
            int m = r + (kg << 3);
            hf[m * HID + col0 + mn] = acc[r] + b1[col0 + mn];
        }
    }
    __syncthreads();

    // ---- LayerNorm(256) + ReLU per row; write A-swizzled f16 ---------------
    {
        const int row = tid >> 4;
        const int sub = tid & 15;
        float s = 0.f, ss = 0.f;
        #pragma unroll
        for (int j = 0; j < 16; ++j) {
            float v = hf[row * HID + sub + 16 * j];
            s += v; ss += v * v;
        }
        #pragma unroll
        for (int m = 1; m < 16; m <<= 1) {
            s  += __shfl_xor(s,  m, 16);
            ss += __shfl_xor(ss, m, 16);
        }
        const float mu  = s * (1.0f / HID);
        const float var = ss * (1.0f / HID) - mu * mu;
        const float inv = rsqrtf(var + LN_EPS);
        #pragma unroll
        for (int j = 0; j < 16; ++j) {
            int col = sub + 16 * j;
            float v = (hf[row * HID + col] - mu) * inv * g[col] + be[col];
            v = fmaxf(v, 0.0f);
            // inverse A-layout: col -> (kk2, lane2, e2) for row `row`
            int kk2   = col >> 5;
            int rem   = col & 31;
            int e2    = (rem & 7) | (((rem >> 4) & 1) << 3);
            int kg2   = (rem >> 3) & 1;
            int lane2 = row + (kg2 << 4);
            h16s[(kk2 * 32 + lane2) * 16 + e2] = (_Float16)v;
        }
    }
    __syncthreads();

    // ---- GEMM2: out = relu(h) @ w2 + b2 ------------------------------------
    const int  n2t = n2 >> 4;                       // 3 or 9 tiles
    const v16h* bW2 = (const v16h*)w2s;
    for (int tcol = wave; tcol < n2t; tcol += 8) {
        v8f acc = {};
        #pragma unroll
        for (int kk = 0; kk < KT2; ++kk) {
            v16h a  = *(const v16h*)&h16s[(kk * 32 + lane) * 16];
            v16h bm = bW2[(kk * n2t + tcol) * 32 + lane];
            acc = __builtin_amdgcn_wmma_f32_16x16x32_f16(
                false, a, false, bm, (short)0, acc, false, false);
        }
        const int col0 = tcol * 16;
        #pragma unroll
        for (int r = 0; r < 8; ++r) {
            int m = r + (kg << 3);
            out[(row0 + m) * n2 + col0 + mn] = acc[r] + b2[col0 + mn];
        }
    }
}

// ---------------------------------------------------------------------------
extern "C" void kernel_launch(void* const* d_in, const int* in_sizes, int n_in,
                              void* d_out, int out_size, void* d_ws, size_t ws_size,
                              hipStream_t stream) {
    const float*     feat    = (const float*)d_in[0];
    const long long* offsets = (const long long*)d_in[1];
    const float*     info    = (const float*)d_in[2];
    const float*     w_info  = (const float*)d_in[3];
    const float*     b_info  = (const float*)d_in[4];
    const float*     w_a1    = (const float*)d_in[5];
    const float*     b_a1    = (const float*)d_in[6];
    const float*     g_a     = (const float*)d_in[7];
    const float*     be_a    = (const float*)d_in[8];
    const float*     w_a2    = (const float*)d_in[9];
    const float*     b_a2    = (const float*)d_in[10];
    const float*     w_o1    = (const float*)d_in[11];
    const float*     b_o1    = (const float*)d_in[12];
    const float*     g_o     = (const float*)d_in[13];
    const float*     be_o    = (const float*)d_in[14];
    const float*     w_o2    = (const float*)d_in[15];
    const float*     b_o2    = (const float*)d_in[16];

    float* out      = (float*)d_out;
    float* features = out;                                  // [1024][512]
    float* pred     = out + BATCH * CH;                     // [1024][48]
    float* offs     = out + BATCH * CH + BATCH * KCLS;      // [1024][144]

    // Workspace layout (all 256B-aligned):
    char* ws = (char*)d_ws;
    float*    x     = (float*)ws;                       // 1024*544*4 = 2228224 B
    size_t off0 = (size_t)BATCH * DIN * 4;
    _Float16* w1a_s = (_Float16*)(ws + off0);           // 544*256*2 = 278528 B
    _Float16* w2a_s = (_Float16*)(ws + off0 + 278528);  // 256*48*2  =  24576 B
    _Float16* w1o_s = (_Float16*)(ws + off0 + 303104);  // 544*256*2 = 278528 B
    _Float16* w2o_s = (_Float16*)(ws + off0 + 581632);  // 256*144*2 =  73728 B

    segmax_kernel<<<BATCH, 256, 0, stream>>>(feat, offsets, features, x);
    info_kernel<<<(BATCH * 32 + 255) / 256, 256, 0, stream>>>(info, w_info, b_info, x);

    swizzle_b_kernel<<<(DIN * HID + 255) / 256, 256, 0, stream>>>(w_a1, w1a_s, DIN, HID);
    swizzle_b_kernel<<<(HID * KCLS + 255) / 256, 256, 0, stream>>>(w_a2, w2a_s, HID, KCLS);
    swizzle_b_kernel<<<(DIN * HID + 255) / 256, 256, 0, stream>>>(w_o1, w1o_s, DIN, HID);
    swizzle_b_kernel<<<(HID * KCLS * 3 + 255) / 256, 256, 0, stream>>>(w_o2, w2o_s, HID, KCLS * 3);

    head_kernel<<<BATCH / 16, 256, 0, stream>>>(x, w1a_s, b_a1, g_a, be_a,
                                                w2a_s, b_a2, pred, KCLS);
    head_kernel<<<BATCH / 16, 256, 0, stream>>>(x, w1o_s, b_o1, g_o, be_o,
                                                w2o_s, b_o2, offs, KCLS * 3);
}
